// CausalMultiHeadAttention_78606491451654
// MI455X (gfx1250) — compile-verified, matrix operand order checked by reading, not run on hardware
//
#include <hip/hip_runtime.h>
#include <hip/hip_bf16.h>

// ---------------------------------------------------------------------------
// Causal MHA forward for MI455X (gfx1250, wave32, WMMA + TDM).
// B=4, S=2048, D=2048, H=16, dk=128. Compute-bound (~0.4 TFLOP vs ~200MB),
// so all heavy math runs on v_wmma_f32_16x16x32_bf16 (f32 accumulate), and
// the attention K-tile staging uses the Tensor Data Mover when available.
// ---------------------------------------------------------------------------

typedef __bf16 bf16_t;
typedef __attribute__((ext_vector_type(16))) __bf16        v16bf;
typedef __attribute__((ext_vector_type(8)))  float         v8f;
typedef __attribute__((ext_vector_type(4)))  unsigned int  u32x4;
typedef __attribute__((ext_vector_type(4)))  int           i32x4;
typedef __attribute__((ext_vector_type(8)))  int           i32x8;

#define BATCH     4
#define SEQ       2048
#define DMODEL    2048
#define NHEADS    16
#define DKH       128

// LDS strides (elements). Chosen so every fragment base is 16B-aligned and
// row-lanes hit distinct banks: 136 elems = 68 dwords (68 mod 64 = 4),
// 40 elems = 20 dwords (20*r mod 64 distinct for r=0..15).
#define KPAD 136
#define VPAD 40
#define PPAD 40

#if defined(__HIP_DEVICE_COMPILE__) && defined(__has_builtin)
#if __has_builtin(__builtin_amdgcn_tensor_load_to_lds) && \
    __has_builtin(__builtin_amdgcn_s_wait_tensorcnt)
#define USE_TDM 1
#endif
#endif

// ---------------------------------------------------------------------------
// Fragment loader (ISA 7.12.2, 16-bit A-matrix 16x32 layout, wave32):
//   lane L: row r = L&15 ; K-half koff = (L>>4)*8
//   v0..v3 : K = koff+{0..7} (packed pairs)   -> one b128 load
//   v4..v7 : K = 16+koff+{0..7}               -> one b128 load
// Serves A from X/Q rows, B from W rows (C=X*W^T), B(=K^T) from K rows,
// A from LDS-P rows, B from LDS-V^T rows. Caller guarantees 16B alignment.
// ---------------------------------------------------------------------------
__device__ __forceinline__ v16bf load_frag16(const bf16_t* p, int ld) {
  const int lane = threadIdx.x & 31;
  const int r    = lane & 15;
  const int koff = (lane >> 4) << 3;   // 0 or 8
  const bf16_t* row = p + (size_t)r * ld + koff;
  union { v16bf v; u32x4 u[2]; } f;
  f.u[0] = *(const u32x4*)(row);
  f.u[1] = *(const u32x4*)(row + 16);
  return f.v;
}

__device__ __forceinline__ v8f wmma_bf16(v16bf a, v16bf b, v8f c) {
  return __builtin_amdgcn_wmma_f32_16x16x32_bf16(
      /*neg_a=*/false, a, /*neg_b=*/false, b,
      /*c_mod=*/(short)0, c, /*reuse_a=*/false, /*reuse_b=*/false);
}

#ifdef USE_TDM
// ---------------------------------------------------------------------------
// Tensor Data Mover: load a 2D bf16 tile [tile_h x tile_w] with global row
// stride `row_stride` (elements) into LDS at byte offset `lds_off`, padding
// the LDS destination every (1<<(pad_icode+1)) dwords by (pad_acode+1) dwords.
// D# per ISA 8.3/8.4. tensor_dim == tile_dim so no OOB path is exercised.
// Issued per-wave (EXEC ignored); caller waits TENSORcnt then barriers.
// ---------------------------------------------------------------------------
__device__ __forceinline__ void tdm_load_tile_bf16(
    unsigned lds_off, const void* gptr, int tile_w, int tile_h,
    int row_stride, int pad_icode, int pad_acode) {
  const unsigned long long ga = (unsigned long long)(size_t)gptr;
  u32x4 g0;
  g0[0] = 1u;                                  // count=1 (valid user D#)
  g0[1] = lds_off;                             // lds_addr (bytes)
  g0[2] = (unsigned)ga;                        // global_addr[31:0]
  g0[3] = (unsigned)((ga >> 32) & 0x01FFFFFFu) // global_addr[56:32]
        | (2u << 30);                          // type=2 ("image")
  i32x8 g1;
  g1[0] = (int)((1u << 16)                     // data_size = 2 bytes
        | (1u << 20)                           // pad_enable
        | (((unsigned)pad_icode & 7u) << 22)   // pad_interval
        | (((unsigned)pad_acode & 127u) << 25)); // pad_amount
  g1[1] = (int)(((unsigned)tile_w & 0xFFFFu) << 16);   // tensor_dim0[15:0]
  g1[2] = (int)((((unsigned)tile_w >> 16) & 0xFFFFu)
        | (((unsigned)tile_h & 0xFFFFu) << 16));       // td0[31:16] | td1[15:0]
  g1[3] = (int)((((unsigned)tile_h >> 16) & 0xFFFFu)
        | (((unsigned)tile_w & 0xFFFFu) << 16));       // td1[31:16] | tile_dim0
  g1[4] = (int)((unsigned)tile_h & 0xFFFFu);           // tile_dim1 ; tile_dim2=0
  g1[5] = (int)(unsigned)row_stride;                   // tensor_dim0_stride[31:0]
  g1[6] = 0;                                           // stride0[47:32]|strd1 lo
  g1[7] = 0;
  const i32x4 gz = {0, 0, 0, 0};
#if __clang_major__ >= 23
  const i32x8 gz8 = {0, 0, 0, 0, 0, 0, 0, 0};
  __builtin_amdgcn_tensor_load_to_lds(g0, g1, gz, gz, gz8, 0);
#else
  __builtin_amdgcn_tensor_load_to_lds(g0, g1, gz, gz, 0);
#endif
}
#endif  // USE_TDM

// ---------------------------------------------------------------------------
// fp32 -> bf16 conversion (grid-stride)
// ---------------------------------------------------------------------------
__global__ void cvt_f32_bf16(const float* __restrict__ in,
                             bf16_t* __restrict__ out, int n) {
  int i = blockIdx.x * blockDim.x + threadIdx.x;
  const int stride = gridDim.x * blockDim.x;
  for (; i < n; i += stride) out[i] = (bf16_t)in[i];
}

// ---------------------------------------------------------------------------
// C[M,N] = A[M,K] * B[N,K]^T   (A,B bf16 row-major; C fp32 or bf16)
// Block: 256 threads = 8 waves. Wave (wm,wn) -> 32x64 tile (2x4 accums).
// Block tile: 128x128. K-loop step 32 -> 8 WMMA/iter/wave.
// ---------------------------------------------------------------------------
template <bool OUT_F32>
__global__ __launch_bounds__(256) void gemm_bf16_nt(
    const bf16_t* __restrict__ A, const bf16_t* __restrict__ B,
    void* __restrict__ Cout, int M, int N, int K) {
  const int w    = threadIdx.x >> 5;
  const int wm   = w & 3;
  const int wn   = w >> 2;
  const int row0 = blockIdx.y * 128 + wm * 32;
  const int col0 = blockIdx.x * 128 + wn * 64;

  v8f acc[2][4];
  for (int mi = 0; mi < 2; ++mi)
    for (int ni = 0; ni < 4; ++ni)
      acc[mi][ni] = (v8f){0.f,0.f,0.f,0.f,0.f,0.f,0.f,0.f};

  const bf16_t* Abase = A + (size_t)row0 * K;
  const bf16_t* Bbase = B + (size_t)col0 * K;

  for (int k0 = 0; k0 < K; k0 += 32) {
    if (k0 + 64 < K) {                 // CDNA5 global_prefetch_b8 of next slab
      __builtin_prefetch(Abase + k0 + 64, 0, 0);
      __builtin_prefetch(Bbase + k0 + 64, 0, 0);
    }
    v16bf a[2], b[4];
    a[0] = load_frag16(Abase + k0, K);
    a[1] = load_frag16(Abase + (size_t)16 * K + k0, K);
    for (int ni = 0; ni < 4; ++ni)
      b[ni] = load_frag16(Bbase + (size_t)(16 * ni) * K + k0, K);
    for (int mi = 0; mi < 2; ++mi)
      for (int ni = 0; ni < 4; ++ni)
        acc[mi][ni] = wmma_bf16(a[mi], b[ni], acc[mi][ni]);
  }

  // C/D layout (ISA 7.12.2): VGPR i, lanes 0-15 -> M=i ; lanes 16-31 -> M=i+8
  const int lane = threadIdx.x & 31;
  const int nl   = lane & 15;
  const int mh   = (lane >> 4) << 3;
  for (int mi = 0; mi < 2; ++mi)
    for (int ni = 0; ni < 4; ++ni)
      for (int i = 0; i < 8; ++i) {
        const int gm = row0 + 16 * mi + i + mh;
        const int gn = col0 + 16 * ni + nl;
        const float v = acc[mi][ni][i];
        if (OUT_F32)
          ((float*)Cout)[(size_t)gm * N + gn] = v;
        else
          ((bf16_t*)Cout)[(size_t)gm * N + gn] = (bf16_t)v;
      }
}

// ---------------------------------------------------------------------------
// Flash attention (causal, online softmax).
// Grid: (SEQ/64, BATCH*NHEADS). Block: 128 threads = 4 waves.
// Each wave owns 16 query rows x dk=128 output (8 f32 accums).
// Per 32-key chunk: TDM stages K into LDS (wave 0) while all threads
// transpose V; QK^T (8 WMMA, B-frags preloaded) -> mask/scale -> online
// softmax (shfl_xor row-reduce in 16-lane halves) -> P to LDS bf16 ->
// PV (8 WMMA, V^T frags preloaded).
// ---------------------------------------------------------------------------
__global__ __launch_bounds__(128) void flash_attn_causal(
    const bf16_t* __restrict__ Q, const bf16_t* __restrict__ Kg,
    const bf16_t* __restrict__ V, bf16_t* __restrict__ Oattn) {
  __shared__ __attribute__((aligned(16))) bf16_t Ksh[32 * KPAD];
  __shared__ __attribute__((aligned(16))) bf16_t Vt[128 * VPAD];
  __shared__ __attribute__((aligned(16))) bf16_t Psh[4 * 16 * PPAD];

  const int wid  = threadIdx.x >> 5;
  const int lane = threadIdx.x & 31;
  const int bh   = blockIdx.y;
  const int b    = bh >> 4;
  const int h    = bh & 15;
  const int qblk = blockIdx.x;
  const int q0   = qblk * 64 + wid * 16;

  const bf16_t* Qbase = Q + ((size_t)(b * SEQ + q0) * DMODEL + h * DKH);
  v16bf qf[4];
  for (int kc = 0; kc < 4; ++kc)
    qf[kc] = load_frag16(Qbase + kc * 32, DMODEL);

  v8f o[8];
  for (int c = 0; c < 8; ++c)
    o[c] = (v8f){0.f,0.f,0.f,0.f,0.f,0.f,0.f,0.f};
  float mrow[8], lrow[8];
  for (int i = 0; i < 8; ++i) { mrow[i] = -1e30f; lrow[i] = 0.f; }

  const int   nl    = lane & 15;
  const int   mh    = (lane >> 4) << 3;
  const float scale = 0.08838834764831845f;   // 1/sqrt(128)
  const int   jend  = qblk * 64 + 64;
  bf16_t* Pme = Psh + wid * (16 * PPAD);

  for (int j0 = 0; j0 < jend; j0 += 32) {
    __syncthreads();   // previous chunk's LDS consumers are done

    const bf16_t* KVsrc_base =
        Kg + ((size_t)(b * SEQ + j0) * DMODEL + h * DKH);
#ifdef USE_TDM
    if (wid == 0) {
      // 32 rows x 128 bf16 from K, LDS rows padded 256B -> 272B:
      // pad_interval code 5 (64 dwords), pad_amount code 3 (4 dwords).
      tdm_load_tile_bf16((unsigned)(size_t)(void*)Ksh, KVsrc_base,
                         /*tile_w=*/DKH, /*tile_h=*/32,
                         /*row_stride=*/DMODEL, /*icode=*/5, /*acode=*/3);
    }
#else
    {
      const unsigned int* Ksrc = (const unsigned int*)KVsrc_base;
      unsigned int* Kdst = (unsigned int*)Ksh;
      for (int i = threadIdx.x; i < 32 * 64; i += 128) {
        const int r = i >> 6, c = i & 63;
        Kdst[r * (KPAD / 2) + c] = Ksrc[(size_t)r * (DMODEL / 2) + c];
      }
    }
#endif
    // V chunk, transposed to [dk][key] so B-fragment key pairs are contiguous
    {
      const bf16_t* Vsrc = V + ((size_t)(b * SEQ + j0) * DMODEL + h * DKH);
      for (int i = threadIdx.x; i < 32 * 128; i += 128) {
        const int r = i >> 7, c = i & 127;
        Vt[c * VPAD + r] = Vsrc[(size_t)r * DMODEL + c];
      }
    }
#ifdef USE_TDM
    if (wid == 0) __builtin_amdgcn_s_wait_tensorcnt(0);
#endif
    __syncthreads();

    // --- scores: S = Q * K^T; preload all 8 B-frags so DS loads pipeline ---
    v16bf kf[2][4];
    for (int t = 0; t < 2; ++t)
      for (int kc = 0; kc < 4; ++kc)
        kf[t][kc] = load_frag16(&Ksh[(16 * t) * KPAD + kc * 32], KPAD);
    v8f s[2];
    s[0] = (v8f){0.f,0.f,0.f,0.f,0.f,0.f,0.f,0.f};
    s[1] = (v8f){0.f,0.f,0.f,0.f,0.f,0.f,0.f,0.f};
    for (int t = 0; t < 2; ++t)
      for (int kc = 0; kc < 4; ++kc)
        s[t] = wmma_bf16(qf[kc], kf[t][kc], s[t]);

    // --- scale + causal mask ---
    for (int t = 0; t < 2; ++t)
      for (int i = 0; i < 8; ++i) {
        const int kgidx = j0 + t * 16 + nl;
        const int qgidx = q0 + i + mh;
        const float v = s[t][i] * scale;
        s[t][i] = (kgidx <= qgidx) ? v : -1e30f;
      }

    // --- online softmax (rows live across 16 lanes of each half-wave) ---
    for (int i = 0; i < 8; ++i) {
      float rm = fmaxf(s[0][i], s[1][i]);
      for (int x = 1; x < 16; x <<= 1) rm = fmaxf(rm, __shfl_xor(rm, x, 16));
      const float mnew  = fmaxf(mrow[i], rm);
      const float alpha = __expf(mrow[i] - mnew);
      const float p0 = __expf(s[0][i] - mnew);
      const float p1 = __expf(s[1][i] - mnew);
      s[0][i] = p0; s[1][i] = p1;
      float rs = p0 + p1;
      for (int x = 1; x < 16; x <<= 1) rs += __shfl_xor(rs, x, 16);
      lrow[i] = lrow[i] * alpha + rs;
      mrow[i] = mnew;
      for (int c = 0; c < 8; ++c) o[c][i] *= alpha;
    }

    // --- P (C-layout f32) -> LDS bf16 row-major for A-fragment reuse ---
    for (int t = 0; t < 2; ++t)
      for (int i = 0; i < 8; ++i)
        Pme[(i + mh) * PPAD + t * 16 + nl] = (bf16_t)s[t][i];
    __syncthreads();

    // --- O += P * V; preload P frag + all 8 V^T frags, then WMMA chain ---
    const v16bf pf = load_frag16(Pme, PPAD);
    v16bf vf[8];
    for (int c = 0; c < 8; ++c)
      vf[c] = load_frag16(&Vt[(c * 16) * VPAD], VPAD);
    for (int c = 0; c < 8; ++c)
      o[c] = wmma_bf16(pf, vf[c], o[c]);
  }

  // --- normalize + store [B,S,H*dk] as bf16 for the output projection ---
  bf16_t* Obase = Oattn + ((size_t)(b * SEQ + q0) * DMODEL + h * DKH);
  for (int i = 0; i < 8; ++i) {
    const float rl = 1.0f / lrow[i];
    for (int c = 0; c < 8; ++c)
      Obase[(size_t)(i + mh) * DMODEL + c * 16 + nl] = (bf16_t)(o[c][i] * rl);
  }
}

// ---------------------------------------------------------------------------
// Host-side orchestration
// ---------------------------------------------------------------------------
extern "C" void kernel_launch(void* const* d_in, const int* in_sizes, int n_in,
                              void* d_out, int out_size, void* d_ws,
                              size_t ws_size, hipStream_t stream) {
  const float* x  = (const float*)d_in[0];
  const float* Wq = (const float*)d_in[1];
  const float* Wk = (const float*)d_in[2];
  const float* Wv = (const float*)d_in[3];
  const float* Wo = (const float*)d_in[4];

  const size_t NX = (size_t)BATCH * SEQ * DMODEL;   // 16.7M elems
  const size_t NW = (size_t)DMODEL * DMODEL;        // 4.2M elems

  bf16_t* p   = (bf16_t*)d_ws;
  bf16_t* xb  = p; p += NX;
  bf16_t* wqb = p; p += NW;
  bf16_t* wkb = p; p += NW;
  bf16_t* wvb = p; p += NW;
  bf16_t* wob = p; p += NW;
  bf16_t* Qb  = p; p += NX;
  bf16_t* Kb  = p; p += NX;
  bf16_t* Vb  = p; p += NX;
  bf16_t* Ab  = p; p += NX;

  cvt_f32_bf16<<<2048, 256, 0, stream>>>(x,  xb,  (int)NX);
  cvt_f32_bf16<<<512,  256, 0, stream>>>(Wq, wqb, (int)NW);
  cvt_f32_bf16<<<512,  256, 0, stream>>>(Wk, wkb, (int)NW);
  cvt_f32_bf16<<<512,  256, 0, stream>>>(Wv, wvb, (int)NW);
  cvt_f32_bf16<<<512,  256, 0, stream>>>(Wo, wob, (int)NW);

  const dim3 gb(DMODEL / 128, (BATCH * SEQ) / 128);   // (16, 64)
  gemm_bf16_nt<false><<<gb, 256, 0, stream>>>(xb, wqb, Qb,
                                              BATCH * SEQ, DMODEL, DMODEL);
  gemm_bf16_nt<false><<<gb, 256, 0, stream>>>(xb, wkb, Kb,
                                              BATCH * SEQ, DMODEL, DMODEL);
  gemm_bf16_nt<false><<<gb, 256, 0, stream>>>(xb, wvb, Vb,
                                              BATCH * SEQ, DMODEL, DMODEL);

  flash_attn_causal<<<dim3(SEQ / 64, BATCH * NHEADS), 128, 0, stream>>>(
      Qb, Kb, Vb, Ab);

  gemm_bf16_nt<true><<<gb, 256, 0, stream>>>(Ab, wob, d_out,
                                             BATCH * SEQ, DMODEL, DMODEL);
}
